// PointNet2PartSegMsg_41274635714773
// MI455X (gfx1250) — matrix operand order repaired and checked by reading, not compiled, not run on hardware
//
#include <hip/hip_runtime.h>
#include <hip/hip_bf16.h>
#include <stdint.h>

typedef _Float16 h16;
typedef _Float16 v16h __attribute__((ext_vector_type(16)));
typedef _Float16 v8h  __attribute__((ext_vector_type(8)));
typedef float    v8f  __attribute__((ext_vector_type(8)));

#define TPB 256

// ---------------- transpose coord (B,6,N) -> xyz0 f32 (B,N,3), feat0 f16 (B,N,6)
__global__ void k_transpose(const float* __restrict__ coord, float* __restrict__ xyz0,
                            h16* __restrict__ feat0, int N, long total) {
  long g = (long)blockIdx.x * blockDim.x + threadIdx.x;
  if (g >= total) return;
  int b = (int)(g / N), p = (int)(g % N);
  const float* c = coord + (size_t)b * 6 * N + p;
  float v0 = c[0], v1 = c[(size_t)N], v2 = c[(size_t)2*N];
  float v3 = c[(size_t)3*N], v4 = c[(size_t)4*N], v5 = c[(size_t)5*N];
  float* xo = xyz0 + (size_t)g * 3;
  xo[0] = v0; xo[1] = v1; xo[2] = v2;
  h16* fo = feat0 + (size_t)g * 6;
  fo[0]=(h16)v0; fo[1]=(h16)v1; fo[2]=(h16)v2; fo[3]=(h16)v3; fo[4]=(h16)v4; fo[5]=(h16)v5;
}

// ---------------- farthest point sampling, one block per batch
__global__ void k_fps(const float* __restrict__ xyz, int N, int S, int* __restrict__ out) {
  __shared__ float dist[2048];
  __shared__ float red[256];
  __shared__ int   redi[256];
  __shared__ int   far_s;
  int b = blockIdx.x;
  int t = threadIdx.x;
  const float* p = xyz + (size_t)b * N * 3;
  for (int j = t; j < N; j += 256) dist[j] = 1e10f;
  if (t == 0) far_s = 0;
  __syncthreads();
  for (int i = 0; i < S; i++) {
    int far = far_s;
    if (t == 0) out[b * S + i] = far;
    float cx = p[far*3], cy = p[far*3+1], cz = p[far*3+2];
    float best = -1.0f; int bi = 0;
    for (int j = t; j < N; j += 256) {
      float dx = p[j*3] - cx, dy = p[j*3+1] - cy, dz = p[j*3+2] - cz;
      float d = dx*dx + dy*dy + dz*dz;
      float nd = fminf(dist[j], d);
      dist[j] = nd;
      if (nd > best) { best = nd; bi = j; }
    }
    red[t] = best; redi[t] = bi;
    __syncthreads();
    for (int s = 128; s > 0; s >>= 1) {
      if (t < s) {
        if (red[t+s] > red[t] || (red[t+s] == red[t] && redi[t+s] < redi[t])) {
          red[t] = red[t+s]; redi[t] = redi[t+s];
        }
      }
      __syncthreads();
    }
    if (t == 0) far_s = redi[0];
    __syncthreads();
  }
}

// ---------------- gather xyz by index: dst (B,S,3) = xyz[idx]
__global__ void k_gatherxyz(const float* __restrict__ xyz, const int* __restrict__ idx,
                            int N, int S, float* __restrict__ dst, long total) {
  long g = (long)blockIdx.x * blockDim.x + threadIdx.x;
  if (g >= total) return;
  int b = (int)(g / S);
  int j = idx[g];
  const float* s = xyz + ((size_t)b * N + j) * 3;
  float* d = dst + (size_t)g * 3;
  d[0] = s[0]; d[1] = s[1]; d[2] = s[2];
}

// ---------------- ball query, one thread per center (single batch, ptrs pre-offset)
__global__ void k_ballquery(const float* __restrict__ xyz, const float* __restrict__ nxyz,
                            int N, int S, int kk, float r2, int* __restrict__ idx) {
  int s = blockIdx.x * blockDim.x + threadIdx.x;
  if (s >= S) return;
  float cx = nxyz[s*3], cy = nxyz[s*3+1], cz = nxyz[s*3+2];
  int cnt = 0;
  int* o = idx + (size_t)s * kk;
  for (int j = 0; j < N && cnt < kk; j++) {
    float dx = xyz[j*3] - cx, dy = xyz[j*3+1] - cy, dz = xyz[j*3+2] - cz;
    if (dx*dx + dy*dy + dz*dz <= r2) { o[cnt++] = j; }
  }
  int first = (cnt > 0) ? o[0] : 0;
  for (; cnt < kk; cnt++) o[cnt] = first;
}

// ---------------- build grouped f16 A matrix [S*kk, Kpad] (single batch)
__global__ void k_group(const int* __restrict__ idx, const float* __restrict__ xyz,
                        const float* __restrict__ nxyz, const h16* __restrict__ feat,
                        int ldf, int C, int kk, int Kpad, h16* __restrict__ A, long total) {
  long e = (long)blockIdx.x * blockDim.x + threadIdx.x;
  if (e >= total) return;
  long row = e / Kpad; int c = (int)(e % Kpad);
  int s = (int)(row / kk);
  int j = idx[row];
  h16 v = (h16)0.0f;
  if (c < 3)            v = (h16)(xyz[j*3 + c] - nxyz[s*3 + c]);
  else if (c - 3 < C)   v = feat[(size_t)j * ldf + (c - 3)];
  A[e] = v;
}

// ---------------- weight convert/pad: f32 [cout,cin] -> f16 [C32o, C32i]
__global__ void k_wconvert(const float* __restrict__ w, int cout, int cin, int C32i,
                           h16* __restrict__ Wp, long total) {
  long e = (long)blockIdx.x * blockDim.x + threadIdx.x;
  if (e >= total) return;
  int r = (int)(e / C32i), c = (int)(e % C32i);
  Wp[e] = (r < cout && c < cin) ? (h16)w[(size_t)r * cin + c] : (h16)0.0f;
}

// ---------------- fold batch norm into scale/bias
__global__ void k_foldbn(const float* __restrict__ b, const float* __restrict__ g,
                         const float* __restrict__ be, const float* __restrict__ m,
                         const float* __restrict__ v, int cout,
                         float* __restrict__ sc, float* __restrict__ bi) {
  int i = blockIdx.x * blockDim.x + threadIdx.x;
  if (i >= cout) return;
  float s = g[i] * rsqrtf(v[i] + 1e-5f);
  sc[i] = s;
  bi[i] = (b[i] - m[i]) * s + be[i];
}

// ---------------- WMMA GEMM, register-tiled: each wave computes 2 M-tiles x NT N-tiles.
// A f16 [M,K] (lda=K), W f16 padded [C32o, ldw=K]; D = A x W^T.
// Epilogue: relu(acc*scale + bias), f16 or f32 out, zero-fills channel padding.
// Block = 128 threads = 4 waves; wave w covers rows (blockIdx.x*4+w)*32 .. +31.
template<int NT>
__global__ void k_gemm_t(const h16* __restrict__ A, int lda,
                         const h16* __restrict__ W, int ldw,
                         void* __restrict__ out, int ldo,
                         int K, int cout, int C32o,
                         const float* __restrict__ scale, const float* __restrict__ bias,
                         int relu, int f32out) {
  int lane = threadIdx.x & 31;
  int wave = threadIdx.x >> 5;
  int half = lane >> 4;     // 0 or 1
  int l = lane & 15;
  int mbase = (blockIdx.x * 4 + wave) * 32;
  int nbase = blockIdx.y * (NT * 16);
  // A fragment rows (16x32 f16 layout): lanes 0-15 K 0-15, lanes 16-31 K 16-31
  const h16* ap0 = A + (size_t)(mbase + l) * lda + half * 16;
  const h16* ap1 = ap0 + (size_t)16 * lda;
  // B fragment (32x16): lane = column n, lanes 0-15 K 0-15, lanes 16-31 K 16-31;
  // B[k][n] = w[n][k] -> contiguous row of padded W.
  const h16* bp[NT];
#pragma unroll
  for (int t = 0; t < NT; t++) {
    int nr = nbase + t * 16 + l;
    if (nr >= C32o) nr = C32o - 1;   // safety clamp (never hit for our shapes)
    bp[t] = W + (size_t)nr * ldw + half * 16;
  }
  v8f acc0[NT], acc1[NT];
#pragma unroll
  for (int t = 0; t < NT; t++) { acc0[t] = {}; acc1[t] = {}; }
  for (int k = 0; k < K; k += 32) {
    union { v16h v; v8h h[2]; } a0, a1;
    a0.h[0] = *(const v8h*)(ap0 + k);
    a0.h[1] = *(const v8h*)(ap0 + k + 8);
    a1.h[0] = *(const v8h*)(ap1 + k);
    a1.h[1] = *(const v8h*)(ap1 + k + 8);
#pragma unroll
    for (int t = 0; t < NT; t++) {
      v16h bv = *(const v16h*)(bp[t] + k);
      acc0[t] = __builtin_amdgcn_wmma_f32_16x16x32_f16(false, a0.v, false, bv,
                                                       (short)0, acc0[t], false, false);
      acc1[t] = __builtin_amdgcn_wmma_f32_16x16x32_f16(false, a1.v, false, bv,
                                                       (short)0, acc1[t], false, false);
    }
  }
  int rb = mbase + half * 8;   // D layout: VGPR r -> M = (half?8:0)+r within tile
#pragma unroll
  for (int t = 0; t < NT; t++) {
    int n = nbase + t * 16 + l;
    if (n >= C32o) continue;
    int nc = (n < cout) ? n : (cout - 1);
    float sc = scale ? scale[nc] : 1.0f;
    float bi = bias  ? bias[nc]  : 0.0f;
#pragma unroll
    for (int r = 0; r < 8; r++) {
      float v0 = acc0[t][r] * sc + bi;
      float v1 = acc1[t][r] * sc + bi;
      if (relu) { v0 = fmaxf(v0, 0.0f); v1 = fmaxf(v1, 0.0f); }
      if (n >= cout) { v0 = 0.0f; v1 = 0.0f; }
      size_t o0 = (size_t)(rb + r) * ldo + n;
      size_t o1 = (size_t)(rb + 16 + r) * ldo + n;
      if (f32out) { ((float*)out)[o0] = v0; ((float*)out)[o1] = v1; }
      else        { ((h16*)out)[o0] = (h16)v0; ((h16*)out)[o1] = (h16)v1; }
    }
  }
}

// ---------------- max over kk rows: out[s, c] = max_j src[(s*kk+j), c]
__global__ void k_maxpool(const h16* __restrict__ src, int ldA, int kk, int cout,
                          h16* __restrict__ out, int ldOut, long total) {
  long e = (long)blockIdx.x * blockDim.x + threadIdx.x;
  if (e >= total) return;
  long s = e / cout; int c = (int)(e % cout);
  const h16* p = src + (size_t)s * kk * ldA + c;
  float m = -1e30f;
  for (int j = 0; j < kk; j++) { float v = (float)p[(size_t)j * ldA]; m = (v > m) ? v : m; }
  out[(size_t)s * ldOut + c] = (h16)m;
}

// ---------------- channel copy into concat buffer (optional row broadcast, f32 or f16 src)
__global__ void k_copych(h16* __restrict__ dst, int ldd, int doff,
                         const void* __restrict__ src, int lds, int C,
                         int is_f32, int divr, long total) {
  long e = (long)blockIdx.x * blockDim.x + threadIdx.x;
  if (e >= total) return;
  long m = e / C; int c = (int)(e % C);
  long sr = m / divr;
  float v = is_f32 ? ((const float*)src)[(size_t)sr * lds + c]
                   : (float)((const h16*)src)[(size_t)sr * lds + c];
  dst[(size_t)m * ldd + doff + c] = (h16)v;
}

__global__ void k_zeroh(h16* __restrict__ p, long n) {
  long e = (long)blockIdx.x * blockDim.x + threadIdx.x;
  if (e < n) p[e] = (h16)0.0f;
}
__global__ void k_zerof(float* __restrict__ p, long n) {
  long e = (long)blockIdx.x * blockDim.x + threadIdx.x;
  if (e < n) p[e] = 0.0f;
}

// ---------------- 3-NN: for each target point find 3 nearest sources, idx3 global rows
__global__ void k_knn3(const float* __restrict__ tgt, const float* __restrict__ src,
                       int Nt, int Ns, long total, int* __restrict__ idx3,
                       float* __restrict__ w3) {
  long m = (long)blockIdx.x * blockDim.x + threadIdx.x;
  if (m >= total) return;
  int b = (int)(m / Nt);
  float tx = tgt[m*3], ty = tgt[m*3+1], tz = tgt[m*3+2];
  const float* s = src + (size_t)b * Ns * 3;
  float d0 = 1e30f, d1 = 1e30f, d2 = 1e30f; int i0 = 0, i1 = 0, i2 = 0;
  for (int j = 0; j < Ns; j++) {
    float dx = s[j*3]-tx, dy = s[j*3+1]-ty, dz = s[j*3+2]-tz;
    float d = dx*dx + dy*dy + dz*dz;
    if (d < d0)      { d2=d1;i2=i1; d1=d0;i1=i0; d0=d;i0=j; }
    else if (d < d1) { d2=d1;i2=i1; d1=d;i1=j; }
    else if (d < d2) { d2=d;i2=j; }
  }
  float w0 = 1.0f/(d0+1e-8f), w1 = 1.0f/(d1+1e-8f), w2 = 1.0f/(d2+1e-8f);
  float ws = w0 + w1 + w2;
  idx3[m*3+0] = b*Ns+i0; idx3[m*3+1] = b*Ns+i1; idx3[m*3+2] = b*Ns+i2;
  w3[m*3+0] = w0/ws; w3[m*3+1] = w1/ws; w3[m*3+2] = w2/ws;
}

// ---------------- weighted 3-NN interpolation into concat buffer
__global__ void k_interp(h16* __restrict__ dst, int ldd, int doff,
                         const h16* __restrict__ src, int lds, int C,
                         const int* __restrict__ idx3, const float* __restrict__ w3,
                         long total) {
  long e = (long)blockIdx.x * blockDim.x + threadIdx.x;
  if (e >= total) return;
  long m = e / C; int c = (int)(e % C);
  float v = w3[m*3+0] * (float)src[(size_t)idx3[m*3+0] * lds + c]
          + w3[m*3+1] * (float)src[(size_t)idx3[m*3+1] * lds + c]
          + w3[m*3+2] * (float)src[(size_t)idx3[m*3+2] * lds + c];
  dst[(size_t)m * ldd + doff + c] = (h16)v;
}

// ---------------- log_softmax over 50 classes per row
__global__ void k_logsoftmax(const float* __restrict__ logits, int ld, int ncls,
                             long rows, float* __restrict__ out) {
  long r = (long)blockIdx.x * blockDim.x + threadIdx.x;
  if (r >= rows) return;
  const float* x = logits + (size_t)r * ld;
  float m = -1e30f;
  for (int c = 0; c < ncls; c++) m = fmaxf(m, x[c]);
  float s = 0.0f;
  for (int c = 0; c < ncls; c++) s += expf(x[c] - m);
  float lse = m + logf(s);
  float* o = out + (size_t)r * ncls;
  for (int c = 0; c < ncls; c++) o[c] = x[c] - lse;
}

// =====================================================================

static inline int a32(int x) { return (x + 31) & ~31; }
static inline int nblk(long total, int tpb) { return (int)((total + tpb - 1) / tpb); }

extern "C" void kernel_launch(void* const* d_in, const int* in_sizes, int n_in,
                              void* d_out, int out_size, void* d_ws, size_t ws_size,
                              hipStream_t stream) {
  (void)in_sizes; (void)n_in; (void)out_size; (void)ws_size;
  const int B = 16, N = 2048;
  const float* coord = (const float*)d_in[0];
  const float* label = (const float*)d_in[1];

  // layer table: leaf index of w (tuple = w,b,g,be,m,v), cin, cout
  // leaf order (jax tree, dict keys sorted): coord, label, fp1, fp2, fp3,
  // head1, head2, sa1(m0,m1,m2), sa2(m0,m1), sa3
  struct LayerD { int leaf, cin, cout; };
  static const LayerD L[26] = {
    {46,9,32},{52,32,32},{58,32,64},        // sa1 mlp0
    {64,9,64},{70,64,64},{76,64,128},       // sa1 mlp1
    {82,9,64},{88,64,96},{94,96,128},       // sa1 mlp2
    {100,323,128},{106,128,128},{112,128,256}, // sa2 mlp0
    {118,323,128},{124,128,196},{130,196,256}, // sa2 mlp1
    {136,515,256},{142,256,512},{148,512,1024},// sa3
    {26,1536,256},{32,256,256},             // fp3
    {14,576,256},{20,256,128},              // fp2
    {2,153,128},{8,128,128},                // fp1
    {38,128,128},                           // head1
    {44,128,50}                             // head2 (linear only)
  };

  // -------- workspace carve-up
  char* ws = (char*)d_ws;
  size_t cur = 0;
  auto alloc = [&](size_t bytes) -> char* {
    cur = (cur + 255) & ~(size_t)255;
    char* p = ws + cur; cur += bytes; return p;
  };
  float* xyz0   = (float*)alloc((size_t)B*N*3*4);
  h16*   feat0h = (h16*)  alloc((size_t)B*N*6*2);
  int*   fps1   = (int*)  alloc((size_t)B*512*4);
  float* xyz1   = (float*)alloc((size_t)B*512*3*4);
  h16*   feat1h = (h16*)  alloc((size_t)B*512*320*2);
  int*   fps2   = (int*)  alloc((size_t)B*128*4);
  float* xyz2   = (float*)alloc((size_t)B*128*3*4);
  h16*   feat2h = (h16*)  alloc((size_t)B*128*512*2);
  h16*   feat3h = (h16*)  alloc((size_t)B*1024*2);
  h16*   feat2p = (h16*)  alloc((size_t)B*128*256*2);
  h16*   feat1p = (h16*)  alloc((size_t)B*512*128*2);
  int*   bq     = (int*)  alloc((size_t)512*128*4);
  int*   knnI   = (int*)  alloc((size_t)B*N*3*4);
  float* knnW   = (float*)alloc((size_t)B*N*3*4);
  float* logits = (float*)alloc((size_t)B*N*64*4);
  h16*   bufA   = (h16*)  alloc((size_t)13*1024*1024);
  h16*   bufB   = (h16*)  alloc((size_t)17*1024*1024);
  h16*   bufC   = (h16*)  alloc((size_t)17*1024*1024);

  h16* Wp[26]; float* Sc[26]; float* Bi[26];
  for (int i = 0; i < 26; i++) {
    int co = a32(L[i].cout), ci = a32(L[i].cin);
    Wp[i] = (h16*)alloc((size_t)co * ci * 2);
    Sc[i] = (float*)alloc((size_t)L[i].cout * 4);
    Bi[i] = (float*)alloc((size_t)L[i].cout * 4);
  }

  // -------- weight prep: pad/convert weights, fold BN
  for (int i = 0; i < 26; i++) {
    int ci = a32(L[i].cin);
    long tot = (long)a32(L[i].cout) * ci;
    k_wconvert<<<nblk(tot, TPB), TPB, 0, stream>>>(
        (const float*)d_in[L[i].leaf], L[i].cout, L[i].cin, ci, Wp[i], tot);
    if (i != 25) {
      int lf = L[i].leaf;
      k_foldbn<<<nblk(L[i].cout, 64), 64, 0, stream>>>(
          (const float*)d_in[lf+1], (const float*)d_in[lf+2], (const float*)d_in[lf+3],
          (const float*)d_in[lf+4], (const float*)d_in[lf+5], L[i].cout, Sc[i], Bi[i]);
    }
  }

  // all GEMM M values are multiples of 128 (block = 4 waves x 32 rows)
  auto gemm = [&](const h16* Ap, int lda, int li, void* out, int ldo, long M,
                  const float* sc, const float* bi, int relu, int f32o) {
    int K = a32(L[li].cin), C32o = a32(L[li].cout);
    if (C32o % 64 == 0) {
      dim3 g((unsigned)(M / 128), (unsigned)(C32o / 64));
      k_gemm_t<4><<<g, 128, 0, stream>>>(Ap, lda, Wp[li], K, out, ldo, K,
                                         L[li].cout, C32o, sc, bi, relu, f32o);
    } else {
      dim3 g((unsigned)(M / 128), (unsigned)(C32o / 32));
      k_gemm_t<2><<<g, 128, 0, stream>>>(Ap, lda, Wp[li], K, out, ldo, K,
                                         L[li].cout, C32o, sc, bi, relu, f32o);
    }
  };

  // -------- stage 0
  k_transpose<<<nblk((long)B*N, TPB), TPB, 0, stream>>>(coord, xyz0, feat0h, N, (long)B*N);

  // -------- SA (multi-scale grouping) helper: per-batch chunks
  auto sa_scale = [&](const float* xyzsrc, int Ns, const float* newxyz, int S,
                      const h16* feat, int ldf, int C, int kk, float rad,
                      int l0, h16* outFeat, int ldOut, int outOff) {
    int Kp = a32(3 + C);
    for (int b = 0; b < B; b++) {
      k_ballquery<<<nblk(S, 64), 64, 0, stream>>>(
          xyzsrc + (size_t)b*Ns*3, newxyz + (size_t)b*S*3, Ns, S, kk, rad*rad, bq);
      long rows = (long)S * kk;
      long tot = rows * Kp;
      k_group<<<nblk(tot, TPB), TPB, 0, stream>>>(
          bq, xyzsrc + (size_t)b*Ns*3, newxyz + (size_t)b*S*3,
          feat + (size_t)b*Ns*ldf, ldf, C, kk, Kp, bufA, tot);
      gemm(bufA, Kp,               l0,   bufB, a32(L[l0].cout),   rows, Sc[l0],   Bi[l0],   1, 0);
      gemm(bufB, a32(L[l0].cout),  l0+1, bufC, a32(L[l0+1].cout), rows, Sc[l0+1], Bi[l0+1], 1, 0);
      gemm(bufC, a32(L[l0+1].cout),l0+2, bufB, a32(L[l0+2].cout), rows, Sc[l0+2], Bi[l0+2], 1, 0);
      int co = L[l0+2].cout;
      long mt = (long)S * co;
      k_maxpool<<<nblk(mt, TPB), TPB, 0, stream>>>(
          bufB, a32(co), kk, co, outFeat + (size_t)b*S*ldOut + outOff, ldOut, mt);
    }
  };

  // -------- SA1: 2048 -> 512 centers, 3 scales -> feat1h (B,512,320)
  k_fps<<<B, 256, 0, stream>>>(xyz0, N, 512, fps1);
  k_gatherxyz<<<nblk((long)B*512, TPB), TPB, 0, stream>>>(xyz0, fps1, N, 512, xyz1, (long)B*512);
  sa_scale(xyz0, N, xyz1, 512, feat0h, 6, 6,  32, 0.1f, 0, feat1h, 320, 0);
  sa_scale(xyz0, N, xyz1, 512, feat0h, 6, 6,  64, 0.2f, 3, feat1h, 320, 64);
  sa_scale(xyz0, N, xyz1, 512, feat0h, 6, 6, 128, 0.4f, 6, feat1h, 320, 192);

  // -------- SA2: 512 -> 128 centers, 2 scales -> feat2h (B,128,512)
  k_fps<<<B, 256, 0, stream>>>(xyz1, 512, 128, fps2);
  k_gatherxyz<<<nblk((long)B*128, TPB), TPB, 0, stream>>>(xyz1, fps2, 512, 128, xyz2, (long)B*128);
  sa_scale(xyz1, 512, xyz2, 128, feat1h, 320, 320,  64, 0.4f,  9, feat2h, 512, 0);
  sa_scale(xyz1, 512, xyz2, 128, feat1h, 320, 320, 128, 0.8f, 12, feat2h, 512, 256);

  // -------- SA3: global MLP over (xyz2, feat2) -> max -> feat3h (B,1024)
  {
    long rows = (long)B * 128;
    int Kp = a32(515); // 544
    k_zeroh<<<nblk(rows*Kp, TPB), TPB, 0, stream>>>(bufA, rows*Kp);
    k_copych<<<nblk(rows*3, TPB), TPB, 0, stream>>>(bufA, Kp, 0, xyz2, 3, 3, 1, 1, rows*3);
    k_copych<<<nblk(rows*512, TPB), TPB, 0, stream>>>(bufA, Kp, 3, feat2h, 512, 512, 0, 1, rows*512);
    gemm(bufA, Kp,  15, bufB, 256,  rows, Sc[15], Bi[15], 1, 0);
    gemm(bufB, 256, 16, bufC, 512,  rows, Sc[16], Bi[16], 1, 0);
    gemm(bufC, 512, 17, bufB, 1024, rows, Sc[17], Bi[17], 1, 0);
    k_maxpool<<<nblk((long)B*1024, TPB), TPB, 0, stream>>>(
        bufB, 1024, 128, 1024, feat3h, 1024, (long)B*1024);
  }

  // -------- FP3: concat(feat2 512, bcast feat3 1024) -> 256 -> 256 -> feat2p
  {
    long rows = (long)B * 128;
    k_copych<<<nblk(rows*512, TPB), TPB, 0, stream>>>(bufA, 1536, 0, feat2h, 512, 512, 0, 1, rows*512);
    k_copych<<<nblk(rows*1024, TPB), TPB, 0, stream>>>(bufA, 1536, 512, feat3h, 1024, 1024, 0, 128, rows*1024);
    gemm(bufA, 1536, 18, bufB,   256, rows, Sc[18], Bi[18], 1, 0);
    gemm(bufB, 256,  19, feat2p, 256, rows, Sc[19], Bi[19], 1, 0);
  }

  // -------- FP2: 3-NN interp feat2p onto xyz1, concat with feat1 -> 256 -> 128 -> feat1p
  {
    long rows = (long)B * 512;
    k_knn3<<<nblk(rows, 64), 64, 0, stream>>>(xyz1, xyz2, 512, 128, rows, knnI, knnW);
    k_copych<<<nblk(rows*320, TPB), TPB, 0, stream>>>(bufA, 576, 0, feat1h, 320, 320, 0, 1, rows*320);
    k_interp<<<nblk(rows*256, TPB), TPB, 0, stream>>>(bufA, 576, 320, feat2p, 256, 256, knnI, knnW, rows*256);
    gemm(bufA, 576, 20, bufB,   256, rows, Sc[20], Bi[20], 1, 0);
    gemm(bufB, 256, 21, feat1p, 128, rows, Sc[21], Bi[21], 1, 0);
  }

  // -------- FP1: skip0 = [label(16), xyz0(3), feat0(6)], interp feat1p (128) -> MLP -> head
  {
    long rows = (long)B * N;
    int Kp = a32(153); // 160
    k_knn3<<<nblk(rows, 64), 64, 0, stream>>>(xyz0, xyz1, N, 512, rows, knnI, knnW);
    k_zeroh<<<nblk(rows*Kp, TPB), TPB, 0, stream>>>(bufA, rows*Kp);
    k_copych<<<nblk(rows*16, TPB), TPB, 0, stream>>>(bufA, Kp, 0,  label,  16, 16, 1, N, rows*16);
    k_copych<<<nblk(rows*3,  TPB), TPB, 0, stream>>>(bufA, Kp, 16, xyz0,   3,  3,  1, 1, rows*3);
    k_copych<<<nblk(rows*6,  TPB), TPB, 0, stream>>>(bufA, Kp, 19, feat0h, 6,  6,  0, 1, rows*6);
    k_interp<<<nblk(rows*128, TPB), TPB, 0, stream>>>(bufA, Kp, 25, feat1p, 128, 128, knnI, knnW, rows*128);
    gemm(bufA, Kp,  22, bufB, 128, rows, Sc[22], Bi[22], 1, 0);
    gemm(bufB, 128, 23, bufC, 128, rows, Sc[23], Bi[23], 1, 0);
    gemm(bufC, 128, 24, bufB, 128, rows, Sc[24], Bi[24], 1, 0);              // head1
    gemm(bufB, 128, 25, logits, 64, rows, nullptr, (const float*)d_in[45], 0, 1); // head2 linear
    k_logsoftmax<<<nblk(rows, 64), 64, 0, stream>>>(logits, 64, 50, rows, (float*)d_out);
    // xyz3 zeros appended after the (B,N,50) output
    k_zerof<<<1, 64, 0, stream>>>((float*)d_out + (size_t)B*N*50, 48);
  }
}
